// GPTQLinear_28862180229399
// MI455X (gfx1250) — matrix operand order repaired and checked by reading, not compiled, not run on hardware
//
#include <hip/hip_runtime.h>
#include <hip/hip_bf16.h>

typedef _Float16 h16_t;
typedef __attribute__((ext_vector_type(16))) _Float16 v16h;
typedef __attribute__((ext_vector_type(8)))  _Float16 v8h;
typedef __attribute__((ext_vector_type(8)))  float    v8f;
typedef __attribute__((ext_vector_type(4)))  int      v4i;

#define IN_F    4096
#define OUT_F   11008
#define GROUPSZ 128
#define NGROUP  (IN_F / GROUPSZ)   /* 32 */
#define WAVES_PER_BLOCK 2
#define TILE_O  16                 /* outputs per wave */

// One wave -> 16 output features, full K reduction.
// A fragment (x tile, 16x32 f16)  : built directly in the wave32 WMMA layout.
// B fragment (W^T tile, 32x16 f16): each lane loads the 8 packed qweight bytes
//   whose 16 nibbles are exactly that lane's B elements, dequants in-register.
__global__ __launch_bounds__(32 * WAVES_PER_BLOCK)
void gptq_wmma_kernel(const h16_t* __restrict__ x,
                      const int*   __restrict__ qw,
                      const h16_t* __restrict__ scales,
                      const h16_t* __restrict__ zeros,
                      const h16_t* __restrict__ bias,
                      h16_t*       __restrict__ out)
{
    const int lane = threadIdx.x & 31;
    const int wave = threadIdx.x >> 5;
    const int n    = lane & 15;   // N (output col in tile) for B/C; M (batch row) for A
    const int h    = lane >> 4;   // which K-half of the fragment this lane holds

    const int o0   = (blockIdx.x * WAVES_PER_BLOCK + wave) * TILE_O;
    const int orow = o0 + n;

    const h16_t* xrow = x + (size_t)(n & 7) * IN_F;     // batch row (rows 8..15 are zero)
    const int*   qrow = qw + (size_t)orow * (IN_F / 2); // packed bytes, 1 byte per int32
    const h16_t* srow = scales + (size_t)orow * NGROUP;
    const h16_t* zrow = zeros  + (size_t)orow * NGROUP;

    v8f acc = {};

    for (int g = 0; g < NGROUP; ++g) {
        const float sf = (float)srow[g];
        const float zs = (float)zrow[g] * sf;   // (q - z)*s == q*s - z*s (f32, like ref)

        // Prefetch this row's qweight four groups (256B/row) ahead; the two
        // half-wave lanes of a row cover both 128B lines of that chunk.
        {
            int pg = g + 4; if (pg > NGROUP - 1) pg = NGROUP - 1;
            __builtin_prefetch(qrow + pg * 64 + 40 * h, 0, 0);
        }

        #pragma unroll
        for (int t = 0; t < 4; ++t) {
            const int kb = g * GROUPSZ + t * 32;

            // ---- A fragment: 16x32 f16, lane holds M=n, K = 16h + {0..15}
            v16h a = {};
            if (n < 8) {
                const v8h alo = *(const v8h*)(xrow + kb + 16 * h);
                const v8h ahi = *(const v8h*)(xrow + kb + 16 * h + 8);
                #pragma unroll
                for (int i = 0; i < 8; ++i) { a[i] = alo[i]; a[i + 8] = ahi[i]; }
            }

            // ---- B fragment: 32x16 f16, lane holds N=n, K = kb + 16h + {0..15}.
            // Those 16 weight columns are nibbles of 8 consecutive packed bytes.
            const int* qp = qrow + (kb >> 1) + 8 * h;
            const v4i q0 = __builtin_nontemporal_load((const v4i*)qp);
            const v4i q1 = __builtin_nontemporal_load((const v4i*)qp + 1);

            v16h b;
            #pragma unroll
            for (int j = 0; j < 8; ++j) {
                const unsigned byteval = (unsigned)(j < 4 ? q0[j] : q1[j - 4]);
                const float lo = (float)(byteval & 15u);         // column 2j   (even)
                const float hi = (float)((byteval >> 4) & 15u);  // column 2j+1 (odd)
                b[2 * j]     = (_Float16)(lo * sf - zs);  // -> v_fma_mixlo_f16
                b[2 * j + 1] = (_Float16)(hi * sf - zs);  // -> v_fma_mixhi_f16
            }

            acc = __builtin_amdgcn_wmma_f32_16x16x32_f16(
                      false, a, false, b, (short)0, acc, false, false);
        }
    }

    // C/D layout: lanes 0-15 hold M=0..7 (VGPR r -> M=r), N = lane. Lanes 16-31
    // hold M=8..15 which are padding rows for batch=8 -> discard.
    const float bf = (float)bias[o0 + n];
    if (h == 0) {
        #pragma unroll
        for (int r = 0; r < 8; ++r)
            out[(size_t)r * OUT_F + o0 + n] = (h16_t)(acc[r] + bf);
    }
}

extern "C" void kernel_launch(void* const* d_in, const int* in_sizes, int n_in,
                              void* d_out, int out_size, void* d_ws, size_t ws_size,
                              hipStream_t stream) {
    (void)in_sizes; (void)n_in; (void)out_size; (void)d_ws; (void)ws_size;
    const h16_t* x      = (const h16_t*)d_in[0];
    const int*   qw     = (const int*)d_in[1];
    const h16_t* scales = (const h16_t*)d_in[2];
    const h16_t* zeros  = (const h16_t*)d_in[3];
    const h16_t* bias   = (const h16_t*)d_in[4];
    h16_t* out = (h16_t*)d_out;

    const int tiles  = OUT_F / TILE_O;           // 688 waves total
    const int blocks = tiles / WAVES_PER_BLOCK;  // 344 workgroups of 64 threads
    gptq_wmma_kernel<<<blocks, 32 * WAVES_PER_BLOCK, 0, stream>>>(
        x, qw, scales, zeros, bias, out);
}
